// RN8_54820962566533
// MI455X (gfx1250) — matrix operand order repaired and checked by reading, not compiled, not run on hardware
//
#include <hip/hip_runtime.h>
#include <hip/hip_bf16.h>

// ---------------------------------------------------------------------------
// Problem constants (from reference): B=256, N=100, V_DIM=2048, Q_DIM=1024
// ---------------------------------------------------------------------------
#define RB      256
#define RN      100
#define VDIM    2048
#define QDIM    1024
#define MBIG    (RB * RN)        // 25600 rows of the big GEMM

typedef __bf16 bf16_t;
typedef __attribute__((ext_vector_type(16))) __bf16 v16bf;
typedef __attribute__((ext_vector_type(8)))  __bf16 v8bf;
typedef __attribute__((ext_vector_type(8)))  float  v8f;

__device__ __forceinline__ unsigned short f2bf(float f) {
    unsigned int u = __float_as_uint(f);
    u += 0x7FFFu + ((u >> 16) & 1u);          // round-to-nearest-even
    return (unsigned short)(u >> 16);
}

// CDNA5 async global->LDS copy (ASYNCcnt-tracked).  lds_addr is the LDS byte
// address (low 32 bits of the generic pointer); offset applies to BOTH sides.
__device__ __forceinline__ void async_g2l_b128(unsigned lds_addr,
                                               const void* gaddr) {
    asm volatile("global_load_async_to_lds_b128 %0, %1, off"
                 :: "v"(lds_addr), "v"(gaddr) : "memory");
}
__device__ __forceinline__ void async_g2l_b128_o16(unsigned lds_addr,
                                                   const void* gaddr) {
    asm volatile("global_load_async_to_lds_b128 %0, %1, off offset:16"
                 :: "v"(lds_addr), "v"(gaddr) : "memory");
}

// ---------------------------------------------------------------------------
// cast f32 -> bf16, 4 elements/thread
// ---------------------------------------------------------------------------
__global__ void cast_bf16x4_kernel(const float* __restrict__ in,
                                   unsigned short* __restrict__ out) {
    int i4 = blockIdx.x * blockDim.x + threadIdx.x;
    float4 t = ((const float4*)in)[i4];
    unsigned int lo = (unsigned int)f2bf(t.x) | ((unsigned int)f2bf(t.y) << 16);
    unsigned int hi = (unsigned int)f2bf(t.z) | ((unsigned int)f2bf(t.w) << 16);
    ((uint2*)out)[i4] = make_uint2(lo, hi);
}

// ---------------------------------------------------------------------------
// transpose + cast: in f32 [K,N] row-major  ->  out bf16 [N,K] row-major
// block (32,8), each block does a 32x32 tile via LDS
// ---------------------------------------------------------------------------
__global__ void transpose_cast_kernel(const float* __restrict__ in,
                                      unsigned short* __restrict__ out,
                                      int K, int N) {
    __shared__ float tile[32][33];
    int n0 = blockIdx.x * 32;
    int k0 = blockIdx.y * 32;
    int tx = threadIdx.x, ty = threadIdx.y;
#pragma unroll
    for (int i = 0; i < 4; ++i)
        tile[ty + i * 8][tx] = in[(size_t)(k0 + ty + i * 8) * N + n0 + tx];
    __syncthreads();
#pragma unroll
    for (int i = 0; i < 4; ++i)
        out[(size_t)(n0 + ty + i * 8) * K + k0 + tx] = f2bf(tile[tx][ty + i * 8]);
}

// ---------------------------------------------------------------------------
// v_sum[b,d] = sum_n v[b,n,d]   (float4 per thread)
// ---------------------------------------------------------------------------
__global__ void vsum_kernel(const float* __restrict__ v, float* __restrict__ vs) {
    int i4 = blockIdx.x * blockDim.x + threadIdx.x;   // b*(VDIM/4) + d4
    int b  = i4 >> 9;                                 // VDIM/4 == 512
    int d4 = i4 & 511;
    const float4* vp = (const float4*)v + (size_t)b * RN * (VDIM / 4) + d4;
    float4 s = make_float4(0.f, 0.f, 0.f, 0.f);
    for (int n = 0; n < RN; ++n) {
        float4 t = vp[(size_t)n * (VDIM / 4)];
        s.x += t.x; s.y += t.y; s.z += t.z; s.w += t.w;
    }
    ((float4*)vs)[i4] = s;
}

// ---------------------------------------------------------------------------
// x[b,n,d] = (RN * v[b,n,d] + vsum[b,d]) * qemb[b,d]  ->  bf16
// ---------------------------------------------------------------------------
__global__ void build_x_kernel(const float* __restrict__ v,
                               const float* __restrict__ vsum,
                               const float* __restrict__ qemb,
                               unsigned short* __restrict__ x) {
    int i4  = blockIdx.x * blockDim.x + threadIdx.x;  // float4 index
    int d4  = i4 & 511;
    int row = i4 >> 9;                                // b*RN + n
    int b   = row / RN;
    float4 vv = ((const float4*)v)[i4];
    float4 vs = ((const float4*)vsum)[(size_t)b * 512 + d4];
    float4 qe = ((const float4*)qemb)[(size_t)b * 512 + d4];
    float x0 = ((float)RN * vv.x + vs.x) * qe.x;
    float x1 = ((float)RN * vv.y + vs.y) * qe.y;
    float x2 = ((float)RN * vv.z + vs.z) * qe.z;
    float x3 = ((float)RN * vv.w + vs.w) * qe.w;
    unsigned int lo = (unsigned int)f2bf(x0) | ((unsigned int)f2bf(x1) << 16);
    unsigned int hi = (unsigned int)f2bf(x2) | ((unsigned int)f2bf(x3) << 16);
    ((uint2*)x)[i4] = make_uint2(lo, hi);
}

// ---------------------------------------------------------------------------
// WMMA bf16 GEMM:  C[M,Nn] = relu(A[M,K] @ B^T(Bt is [Nn,K]) + bias[Nn])
// Block tile 128x128, BK=32, 256 threads = 8 waves arranged 4(M) x 2(N);
// each wave computes 32x64 = 2x4 tiles of 16x16 with v_wmma_f32_16x16x32_bf16.
// Staging: GLOBAL_LOAD_ASYNC_TO_LDS_B128 (ASYNCcnt), 2-deep pipeline over a
// double-buffered LDS tile; s_wait_asynccnt retires the older tile in order.
// ---------------------------------------------------------------------------
#define BM   128
#define BN   128
#define BK   32
#define KPAD 8
#define BSTR (BK + KPAD)   // 40 halves = 80 bytes LDS row stride

__global__ __launch_bounds__(256)
void gemm_bf16_wmma(const unsigned short* __restrict__ A,   // [M,K] bf16
                    const unsigned short* __restrict__ Bt,  // [Nn,K] bf16
                    const float* __restrict__ bias,         // [Nn]
                    float* __restrict__ C,                   // [M,Nn]
                    int M, int Nn, int K) {
    __shared__ __align__(16) unsigned short Abuf[2][BM][BSTR];
    __shared__ __align__(16) unsigned short Bbuf[2][BN][BSTR];

    const int tid  = threadIdx.x;
    const int lane = tid & 31;
    const int wave = tid >> 5;
    const int wm   = wave >> 1;     // 0..3
    const int wn   = wave & 1;      // 0..1
    const int l    = lane & 15;
    const int p    = lane >> 4;     // half-wave select

    const int bn0 = blockIdx.x * BN;
    const int bm0 = blockIdx.y * BM;

    // staging: each thread moves 16 halves (32B) of A and of B per K-step,
    // as two b128 async ops with identical LDS/global relative offsets.
    const int srow = tid >> 1;            // 0..127
    const int scol = (tid & 1) * 16;      // half offset 0 or 16

    const unsigned short* gA = A  + (size_t)(bm0 + srow) * K + scol;
    const unsigned short* gB = Bt + (size_t)(bn0 + srow) * K + scol;

    // LDS byte addresses (generic pointer low 32 bits == LDS offset)
    unsigned ldsA[2], ldsB[2];
    ldsA[0] = (unsigned)(size_t)&Abuf[0][srow][scol];
    ldsA[1] = (unsigned)(size_t)&Abuf[1][srow][scol];
    ldsB[0] = (unsigned)(size_t)&Bbuf[0][srow][scol];
    ldsB[1] = (unsigned)(size_t)&Bbuf[1][srow][scol];

    const int ksteps = K / BK;

    auto issue_tile = [&](int kt, int b) {
        const unsigned short* pa = gA + (size_t)kt * BK;
        const unsigned short* pb = gB + (size_t)kt * BK;
        async_g2l_b128(ldsA[b], pa);
        async_g2l_b128_o16(ldsA[b], pa);
        async_g2l_b128(ldsB[b], pb);
        async_g2l_b128_o16(ldsB[b], pb);
    };

    issue_tile(0, 0);
    if (ksteps > 1) issue_tile(1, 1);

    v8f acc[2][4];
    {
        v8f z = {0.f, 0.f, 0.f, 0.f, 0.f, 0.f, 0.f, 0.f};
#pragma unroll
        for (int i = 0; i < 2; ++i)
#pragma unroll
            for (int j = 0; j < 4; ++j) acc[i][j] = z;
    }

    int buf = 0;
    for (int kt = 0; kt < ksteps; ++kt) {
        // retire tile kt (async loads complete in order); tile kt+1 may
        // remain in flight (4 ops) and keeps overlapping with compute.
        if (kt + 1 < ksteps) {
            asm volatile("s_wait_asynccnt 0x4" ::: "memory");
        } else {
            asm volatile("s_wait_asynccnt 0x0" ::: "memory");
        }
        __syncthreads();                 // tile kt visible to all waves

        if (kt + 2 < ksteps) {           // L2 prefetch two tiles ahead
            __builtin_prefetch(gA + (size_t)(kt + 2) * BK, 0, 0);
            __builtin_prefetch(gB + (size_t)(kt + 2) * BK, 0, 0);
        }

        // --- fragments from LDS (ISA VGPR layouts, wave32) ---
        v16bf afrag[2], bfrag[4];
#pragma unroll
        for (int tm = 0; tm < 2; ++tm) {
            const unsigned short* ap = &Abuf[buf][wm * 32 + tm * 16 + l][p * 8];
            v8bf lo = *(const v8bf*)ap;           // K = p*8 .. p*8+7
            v8bf hi = *(const v8bf*)(ap + 16);    // K = 16+p*8 .. 23+p*8
            afrag[tm] = __builtin_shufflevector(lo, hi,
                0, 1, 2, 3, 4, 5, 6, 7, 8, 9, 10, 11, 12, 13, 14, 15);
        }
#pragma unroll
        for (int tn = 0; tn < 4; ++tn) {
            const unsigned short* bp = &Bbuf[buf][wn * 64 + tn * 16 + l][p * 8];
            v8bf lo = *(const v8bf*)bp;           // K = p*8 .. p*8+7
            v8bf hi = *(const v8bf*)(bp + 16);    // K = 16+p*8 .. 23+p*8
            bfrag[tn] = __builtin_shufflevector(lo, hi,
                0, 1, 2, 3, 4, 5, 6, 7, 8, 9, 10, 11, 12, 13, 14, 15);
        }

#pragma unroll
        for (int tm = 0; tm < 2; ++tm)
#pragma unroll
            for (int tn = 0; tn < 4; ++tn)
                acc[tm][tn] = __builtin_amdgcn_wmma_f32_16x16x32_bf16(
                    false, afrag[tm], false, bfrag[tn],
                    (short)0, acc[tm][tn], false, false);

        __syncthreads();                 // all waves done reading lds[buf]
        if (kt + 2 < ksteps)
            issue_tile(kt + 2, buf);     // refill the buffer just freed
        buf ^= 1;
    }

    // --- epilogue: bias + ReLU, f32 store.  C tile layout: lane l = column,
    // vgpr i = row (p selects rows 0-7 / 8-15 of the 16x16 tile) ---
#pragma unroll
    for (int tm = 0; tm < 2; ++tm) {
        const int row0 = bm0 + wm * 32 + tm * 16 + p * 8;
#pragma unroll
        for (int tn = 0; tn < 4; ++tn) {
            const int col = bn0 + wn * 64 + tn * 16 + l;
            const float bb = bias[col];
#pragma unroll
            for (int i = 0; i < 8; ++i) {
                float r = acc[tm][tn][i] + bb;
                r = fmaxf(r, 0.0f);
                C[(size_t)(row0 + i) * Nn + col] = r;
            }
        }
    }
}

// ---------------------------------------------------------------------------
// launch
// ---------------------------------------------------------------------------
extern "C" void kernel_launch(void* const* d_in, const int* in_sizes, int n_in,
                              void* d_out, int out_size, void* d_ws, size_t ws_size,
                              hipStream_t stream) {
    const float* v  = (const float*)d_in[0];
    const float* q  = (const float*)d_in[1];
    const float* W3 = (const float*)d_in[2];
    const float* b3 = (const float*)d_in[3];
    const float* W2 = (const float*)d_in[4];
    const float* b2 = (const float*)d_in[5];
    float* out = (float*)d_out;

    char* ws = (char*)d_ws;
    // workspace layout
    const size_t off_qbf  = 0;                                    // 256*1024 bf16
    const size_t off_w3t  = off_qbf  + (size_t)RB * QDIM * 2;     // 2048x1024 bf16
    const size_t off_w2t  = off_w3t  + (size_t)VDIM * QDIM * 2;   // 2048x2048 bf16
    const size_t off_qemb = off_w2t  + (size_t)VDIM * VDIM * 2;   // 256x2048 f32
    const size_t off_vsum = off_qemb + (size_t)RB * VDIM * 4;     // 256x2048 f32
    const size_t off_x    = off_vsum + (size_t)RB * VDIM * 4;     // 25600x2048 bf16

    unsigned short* qbf  = (unsigned short*)(ws + off_qbf);
    unsigned short* w3t  = (unsigned short*)(ws + off_w3t);
    unsigned short* w2t  = (unsigned short*)(ws + off_w2t);
    float*          qemb = (float*)(ws + off_qemb);
    float*          vsum = (float*)(ws + off_vsum);
    unsigned short* xbf  = (unsigned short*)(ws + off_x);

    // 1) q -> bf16                        (256*1024 / 4 / 256 = 256 blocks)
    cast_bf16x4_kernel<<<256, 256, 0, stream>>>(q, qbf);

    // 2) W3 [QDIM,VDIM] -> W3T bf16 [VDIM,QDIM]
    transpose_cast_kernel<<<dim3(VDIM / 32, QDIM / 32), dim3(32, 8), 0, stream>>>(
        W3, w3t, QDIM, VDIM);

    // 3) W2 [VDIM,VDIM] -> W2T bf16 [VDIM,VDIM]
    transpose_cast_kernel<<<dim3(VDIM / 32, VDIM / 32), dim3(32, 8), 0, stream>>>(
        W2, w2t, VDIM, VDIM);

    // 4) q_emb = relu(q @ W3 + b3)   M=256, N=2048, K=1024
    gemm_bf16_wmma<<<dim3(VDIM / BN, RB / BM), 256, 0, stream>>>(
        qbf, w3t, b3, qemb, RB, VDIM, QDIM);

    // 5) v_sum                         (256*2048 / 4 / 256 = 512 blocks)
    vsum_kernel<<<512, 256, 0, stream>>>(v, vsum);

    // 6) x = (100*v + v_sum) * q_emb  -> bf16   (25600*2048/4/256 = 51200)
    build_x_kernel<<<51200, 256, 0, stream>>>(v, vsum, qemb, xbf);

    // 7) out = relu(x @ W2 + b2)   M=25600, N=2048, K=2048
    gemm_bf16_wmma<<<dim3(VDIM / BN, MBIG / BM), 256, 0, stream>>>(
        xbf, w2t, b2, out, MBIG, VDIM, VDIM);
}
